// deformable_dense_Context_43946105373417
// MI455X (gfx1250) — matrix operand order repaired and checked by reading, not compiled, not run on hardware
//
#include <hip/hip_runtime.h>

// ---------------- problem constants ----------------
#define B_N   2
#define C_CH  128
#define H_IN  96
#define W_IN  96
#define KKR   9            // 3x3 taps
#define KDIM  1152         // C*KK reduction depth
#define KT_TOT 36          // 1152 / 32
#define KT_HALF 18         // half-K panel (576 rows) to bound LDS
#define KHALF 576
#define TN    64           // 8x8 spatial tile = GEMM N
#define PKROW 592          // padded panel row stride (bf16 elems), 16-elem aligned
#define EPS_BN 1e-5f

typedef __attribute__((ext_vector_type(16))) __bf16 v16bf;
typedef __attribute__((ext_vector_type(8)))  float  v8f;

static __device__ __forceinline__ unsigned short f2bf(float f) {
  unsigned u = __builtin_bit_cast(unsigned, f);
  u += 0x7fffu + ((u >> 16) & 1u);           // round-to-nearest-even
  return (unsigned short)(u >> 16);
}

// -------------------------------------------------------------------------
// Pre-swizzle conv weights (O,C,3,3) f32 -> bf16 WMMA A-fragment order.
// A-frag (16-bit 16x32, contiguous k order used here): lane L: M = 16*mt + L%16, kgroup = L/16,
// element e: k = 32*kt + 16*kgroup + e.  K index = c*9 + kk.
// -------------------------------------------------------------------------
__global__ void prep_w_frags(const float* __restrict__ src,
                             unsigned short* __restrict__ dst,
                             int Oreal, int total) {
  int t = blockIdx.x * 256 + threadIdx.x;
  if (t >= total) return;
  int e    = t & 15;
  int lane = (t >> 4) & 31;
  int frag = t >> 9;
  int kt = frag % KT_TOT;
  int mt = frag / KT_TOT;
  int o  = mt * 16 + (lane & 15);
  int kg = lane >> 4;
  int kl = 16 * kg + e;
  int K  = kt * 32 + kl;
  int c  = K / KKR;
  int kk = K - KKR * c;
  float v = (o < Oreal) ? src[((size_t)o * C_CH + c) * KKR + kk] : 0.0f;
  dst[t] = f2bf(v);
}

// -------------------------------------------------------------------------
// Fused deformable block: offset-conv (WMMA) -> bilinear sample -> main
// GEMM (WMMA) -> BN + ReLU + residual.  One 8x8 pixel tile per workgroup.
// -------------------------------------------------------------------------
__global__ __launch_bounds__(256, 1)
void fused_block(const float* __restrict__ xin, float* __restrict__ out,
                 const unsigned short* __restrict__ offw_fr,
                 const float* __restrict__ off_b,
                 const unsigned short* __restrict__ w_fr,
                 const float* __restrict__ gbn, const float* __restrict__ bbn,
                 const float* __restrict__ mbn, const float* __restrict__ vbn,
                 int dil) {
  __shared__ __attribute__((aligned(32))) unsigned short panel[TN * PKROW]; // 75,776 B
  __shared__ float  offarr[18 * TN];   // y/x offsets, channel-major
  __shared__ float  mskarr[9 * TN];    // sigmoid mask
  __shared__ int4   scoord[9 * TN];    // 4 clamped bilinear corner indices
  __shared__ float4 swt[9 * TN];       // 4 corner weights (valid*mask folded)

  const int tid  = threadIdx.x;
  const int lane = tid & 31;
  const int wv   = tid >> 5;
  const int b    = blockIdx.z;
  const int tx0  = blockIdx.x * 8;
  const int ty0  = blockIdx.y * 8;
  const int pad  = dil;                        // dil*(K-1)/2 with K=3
  const float* xb = xin + (size_t)b * C_CH * H_IN * W_IN;

  // ---------------- offset conv: 2 Mtiles x 4 Ntiles, one tile per wave ---
  const int omt = wv & 1, ont = wv >> 1;
  v8f oacc = {0.f, 0.f, 0.f, 0.f, 0.f, 0.f, 0.f, 0.f};
  const int kg = lane >> 4;

  for (int h = 0; h < 2; ++h) {
    // build im2col panel for channels [64h, 64h+64)
    for (int idx = tid; idx < TN * KHALF; idx += 256) {
      int n  = idx / KHALF;
      int kl = idx - n * KHALF;
      int K  = KHALF * h + kl;
      int c  = K / KKR;
      int kk = K - KKR * c;
      int yy = ty0 + (n >> 3) - pad + (kk / 3) * dil;
      int xx = tx0 + (n & 7)  - pad + (kk % 3) * dil;
      float v = 0.0f;
      if (yy >= 0 && yy < H_IN && xx >= 0 && xx < W_IN)
        v = xb[(c * H_IN + yy) * W_IN + xx];
      panel[n * PKROW + kl] = f2bf(v);
    }
    __syncthreads();

    const unsigned short* afr =
        offw_fr + ((size_t)(omt * KT_TOT + h * KT_HALF) * 32 + lane) * 16;
    const int ncol = ont * 16 + (lane & 15);
    for (int kt = 0; kt < KT_HALF; ++kt) {
      v16bf a   = *(const v16bf*)(afr + (size_t)kt * 32 * 16);
      v16bf bfr = *(const v16bf*)&panel[ncol * PKROW + kt * 32 + 16 * kg];
      oacc = __builtin_amdgcn_wmma_f32_16x16x32_bf16(
          false, a, false, bfr, (short)0, oacc, false, false);
    }
    __syncthreads();   // done reading panel before next half rebuilds it
  }

  // scatter offset-conv results: channels 0..17 offsets, 18..26 sigmoid mask
  {
    const int ncol  = ont * 16 + (lane & 15);
    const int obase = omt * 16 + 8 * kg;
#pragma unroll
    for (int r = 0; r < 8; ++r) {
      int o = obase + r;
      if (o < 27) {
        float val = oacc[r] + off_b[o];
        if (o < 18) offarr[o * TN + ncol] = val;
        else        mskarr[(o - 18) * TN + ncol] = 1.0f / (1.0f + __expf(-val));
      }
    }
  }
  __syncthreads();

  // bilinear metadata, computed once per (kk, n), reused by all 128 channels
  for (int idx = tid; idx < 9 * TN; idx += 256) {
    int kk = idx / TN;
    int n  = idx - kk * TN;
    float gy = (float)(ty0 + (n >> 3) - pad + (kk / 3) * dil);
    float gx = (float)(tx0 + (n & 7)  - pad + (kk % 3) * dil);
    float py = gy + offarr[(2 * kk + 0) * TN + n];
    float px = gx + offarr[(2 * kk + 1) * TN + n];
    float mv = mskarr[idx];
    float y0f = floorf(py), x0f = floorf(px);
    float ly = py - y0f, lx = px - x0f;
    int y0 = (int)y0f, x0 = (int)x0f;
    int y1 = y0 + 1,  x1 = x0 + 1;
    float vy0 = (y0 >= 0 && y0 < H_IN) ? 1.f : 0.f;
    float vy1 = (y1 >= 0 && y1 < H_IN) ? 1.f : 0.f;
    float vx0 = (x0 >= 0 && x0 < W_IN) ? 1.f : 0.f;
    float vx1 = (x1 >= 0 && x1 < W_IN) ? 1.f : 0.f;
    int y0c = min(max(y0, 0), H_IN - 1), y1c = min(max(y1, 0), H_IN - 1);
    int x0c = min(max(x0, 0), W_IN - 1), x1c = min(max(x1, 0), W_IN - 1);
    scoord[idx] = make_int4(y0c * W_IN + x0c, y0c * W_IN + x1c,
                            y1c * W_IN + x0c, y1c * W_IN + x1c);
    swt[idx] = make_float4((1.f - ly) * (1.f - lx) * vy0 * vx0 * mv,
                           (1.f - ly) * lx         * vy0 * vx1 * mv,
                           ly         * (1.f - lx) * vy1 * vx0 * mv,
                           ly         * lx         * vy1 * vx1 * mv);
  }
  __syncthreads();

  // ---------------- main GEMM: wave -> 4 Mtiles x 1 Ntile -----------------
  const int nt  = wv & 3;
  const int mtb = (wv >> 2) * 4;
  v8f macc[4];
#pragma unroll
  for (int i = 0; i < 4; ++i) macc[i] = (v8f){0.f,0.f,0.f,0.f,0.f,0.f,0.f,0.f};

  for (int h = 0; h < 2; ++h) {
    // build bilinearly-sampled panel for channels [64h, 64h+64)
    for (int idx = tid; idx < TN * KHALF; idx += 256) {
      int n  = idx / KHALF;
      int kl = idx - n * KHALF;
      int K  = KHALF * h + kl;
      int c  = K / KKR;
      int kk = K - KKR * c;
      int si = kk * TN + n;
      int4   sc = scoord[si];
      float4 wt = swt[si];
      const float* xp = xb + (size_t)c * H_IN * W_IN;
      float v = wt.x * xp[sc.x] + wt.y * xp[sc.y] +
                wt.z * xp[sc.z] + wt.w * xp[sc.w];
      panel[n * PKROW + kl] = f2bf(v);
    }
    __syncthreads();

    const int ncol = nt * 16 + (lane & 15);
    for (int kt = 0; kt < KT_HALF; ++kt) {
      v16bf bfr = *(const v16bf*)&panel[ncol * PKROW + kt * 32 + 16 * kg];
      if (kt + 1 < KT_HALF)   // stream next A slice through GL2
        __builtin_prefetch(
            w_fr + ((size_t)(mtb * KT_TOT + h * KT_HALF + kt + 1) * 32 + lane) * 16,
            0, 0);
#pragma unroll
      for (int i = 0; i < 4; ++i) {
        const unsigned short* ap =
            w_fr + ((size_t)((mtb + i) * KT_TOT + h * KT_HALF + kt) * 32 + lane) * 16;
        v16bf a = *(const v16bf*)ap;
        macc[i] = __builtin_amdgcn_wmma_f32_16x16x32_bf16(
            false, a, false, bfr, (short)0, macc[i], false, false);
      }
    }
    __syncthreads();
  }

  // ---------------- epilogue: BN + ReLU + residual ------------------------
  {
    const int ncol = nt * 16 + (lane & 15);
    const int yy = ty0 + (ncol >> 3);
    const int xx = tx0 + (ncol & 7);
    float* ob = out + (size_t)b * C_CH * H_IN * W_IN;
#pragma unroll
    for (int i = 0; i < 4; ++i) {
      int obch = (mtb + i) * 16 + 8 * kg;
#pragma unroll
      for (int r = 0; r < 8; ++r) {
        int o = obch + r;
        float sc  = gbn[o] * rsqrtf(vbn[o] + EPS_BN);
        float val = (macc[i][r] - mbn[o]) * sc + bbn[o];
        val = fmaxf(val, 0.0f);
        size_t off = ((size_t)o * H_IN + yy) * W_IN + xx;
        ob[off] = val + xb[off];
      }
    }
  }
}

// -------------------------------------------------------------------------
extern "C" void kernel_launch(void* const* d_in, const int* in_sizes, int n_in,
                              void* d_out, int out_size, void* d_ws, size_t ws_size,
                              hipStream_t stream) {
  (void)in_sizes; (void)n_in; (void)out_size; (void)ws_size;
  const float* hu = (const float*)d_in[0];

  char* ws = (char*)d_ws;
  size_t off = 0;
  unsigned short* owfr[3];
  unsigned short* wfr[3];
  for (int i = 0; i < 3; ++i) { owfr[i] = (unsigned short*)(ws + off); off += (size_t)2 * KT_TOT * 512 * 2; }
  for (int i = 0; i < 3; ++i) { wfr[i]  = (unsigned short*)(ws + off); off += (size_t)8 * KT_TOT * 512 * 2; }
  float* a1 = (float*)(ws + off); off += (size_t)B_N * C_CH * H_IN * W_IN * 4;
  float* a2 = (float*)(ws + off);

  // weight pre-swizzle into WMMA fragment order (bf16)
  for (int i = 0; i < 3; ++i) {
    const float* offw  = (const float*)d_in[1 + 7 * i + 0];
    const float* wmain = (const float*)d_in[1 + 7 * i + 2];
    int tot_o = 2 * KT_TOT * 512;
    prep_w_frags<<<(tot_o + 255) / 256, 256, 0, stream>>>(offw, owfr[i], 27, tot_o);
    int tot_m = 8 * KT_TOT * 512;
    prep_w_frags<<<(tot_m + 255) / 256, 256, 0, stream>>>(wmain, wfr[i], 128, tot_m);
  }

  dim3 grid(12, 12, B_N), blk(256);
  const float* x = hu;
  float* outs[3] = {a1, a2, (float*)d_out};
  int dils[3] = {2, 4, 8};
  for (int i = 0; i < 3; ++i) {
    const float* ob  = (const float*)d_in[1 + 7 * i + 1];
    const float* g   = (const float*)d_in[1 + 7 * i + 3];
    const float* bb  = (const float*)d_in[1 + 7 * i + 4];
    const float* mm  = (const float*)d_in[1 + 7 * i + 5];
    const float* vv  = (const float*)d_in[1 + 7 * i + 6];
    fused_block<<<grid, blk, 0, stream>>>(x, outs[i], owfr[i], ob, wfr[i],
                                          g, bb, mm, vv, dils[i]);
    x = outs[i];
  }
}
